// RefinedSamplingBlock_7584912245252
// MI455X (gfx1250) — matrix operand order repaired and checked by reading, not compiled, not run on hardware
//
#include <hip/hip_runtime.h>
#include <hip/hip_bf16.h>

typedef __attribute__((ext_vector_type(16))) _Float16 v16h;
typedef __attribute__((ext_vector_type(8)))  _Float16 v8h;
typedef __attribute__((ext_vector_type(8)))  float    v8f;

#define DD 768
#define MS 53
#define NPATCH 2809      // 53*53
#define NK 196
#define NKPAD 208
#define NTOP 100
#define NTILE 48         // 768/16
#define OUTROWS 297      // 197 + 100

__device__ __forceinline__ v8f wmma16(v16h a, v16h b, v8f c) {
    return __builtin_amdgcn_wmma_f32_16x16x32_f16(false, a, false, b, (short)0, c, false, false);
}

// A-fragment: 16x32 (MxK) f16 from row-major matrix, row stride ld (elements).
// ISA layout: lanes 0-15 -> M=lane, K pairs {0..7,16..23}; lanes 16-31 -> K {8..15,24..31}.
__device__ __forceinline__ v16h load_frag_a(const _Float16* base, int ld, int lane) {
    const int half = lane >> 4;
    const int m    = lane & 15;
    const _Float16* row = base + m * ld + half * 8;
    v8h lo = *(const v8h*)(row);        // K = half*8 .. half*8+7
    v8h hi = *(const v8h*)(row + 16);   // K = 16+half*8 .. +7
    v16h r;
#pragma unroll
    for (int i = 0; i < 8; ++i) { r[i] = lo[i]; r[i + 8] = hi[i]; }
    return r;
}

// B-fragment: 32x16 (KxN) f16; source stored N-major (transposed), row stride ld.
// ISA layout: lane = N (mod 16); lanes 0-15 hold K {0..7,16..23}, lanes 16-31 hold K {8..15,24..31}.
__device__ __forceinline__ v16h load_frag_b(const _Float16* bt, int ld, int lane) {
    const int half = lane >> 4;
    const int n    = lane & 15;
    const _Float16* col = bt + n * ld + half * 8;
    v8h lo = *(const v8h*)(col);
    v8h hi = *(const v8h*)(col + 16);
    v16h r;
#pragma unroll
    for (int i = 0; i < 8; ++i) { r[i] = lo[i]; r[i + 8] = hi[i]; }
    return r;
}

// ---------- kernel 1: transpose+convert the three 768x768 weights to f16 ----------
__global__ void __launch_bounds__(256) cvt_weights_kernel(
        const float* __restrict__ Wq, const float* __restrict__ Wk,
        const float* __restrict__ Wo,
        _Float16* __restrict__ WqT, _Float16* __restrict__ WkT,
        _Float16* __restrict__ WoT) {
    int idx = blockIdx.x * 256 + threadIdx.x;
    if (idx >= DD * DD) return;
    int k = idx / DD, n = idx % DD;             // coalesced read of src row k
    const float* src; _Float16* dst;
    if (blockIdx.y == 0)      { src = Wq; dst = WqT; }
    else if (blockIdx.y == 1) { src = Wk; dst = WkT; }
    else                      { src = Wo; dst = WoT; }
    dst[n * DD + k] = (_Float16)src[k * DD + n];
}

// ---------- kernel 2: copy vit_features into out[:, :197, :] ----------
__global__ void __launch_bounds__(256) copy_vit_kernel(
        const float* __restrict__ vit, float* __restrict__ out, int Bn) {
    size_t idx = (size_t)blockIdx.x * 256 + threadIdx.x;
    size_t total = (size_t)Bn * 197 * DD;
    if (idx >= total) return;
    size_t b = idx / (197 * DD), r = idx % (197 * DD);
    out[b * (size_t)(OUTROWS * DD) + r] = vit[idx];
}

// ---------- kernel 3: k = LN(vit[:,1:] @ Wk + bk) -> f16, padded to 208 rows ----------
__global__ void __launch_bounds__(256) kproj_kernel(
        const float* __restrict__ vit, const _Float16* __restrict__ WkT,
        const float* __restrict__ bk, const float* __restrict__ lnw,
        const float* __restrict__ lnb, _Float16* __restrict__ k16) {
    __shared__ _Float16 sA[16 * DD];
    __shared__ _Float16 sC[16 * DD];
    const int tid = threadIdx.x, wave = tid >> 5, lane = tid & 31;
    const int jt = blockIdx.x, b = blockIdx.y;

    for (int m = 0; m < 16; ++m) {
        int j = jt * 16 + m;
        const float* src = vit + ((size_t)b * 197 + 1 + j) * DD;
        for (int d = tid; d < DD; d += 256)
            sA[m * DD + d] = (j < NK) ? (_Float16)src[d] : (_Float16)0.0f;
    }
    __syncthreads();

    for (int nt = wave; nt < NTILE; nt += 8) {
        v8f acc = {};
        const _Float16* bt = WkT + (size_t)nt * 16 * DD;
        for (int kc = 0; kc < DD; kc += 32)
            acc = wmma16(load_frag_a(sA + kc, DD, lane), load_frag_b(bt + kc, DD, lane), acc);
        const int half = lane >> 4, n = lane & 15;
        const float bias = bk[nt * 16 + n];
#pragma unroll
        for (int r = 0; r < 8; ++r)
            sC[(r + 8 * half) * DD + nt * 16 + n] = (_Float16)(acc[r] + bias);
    }
    __syncthreads();

    for (int m = wave * 2; m < wave * 2 + 2; ++m) {
        const _Float16* row = sC + m * DD;
        float s = 0.f;
        for (int d = lane; d < DD; d += 32) s += (float)row[d];
#pragma unroll
        for (int off = 16; off > 0; off >>= 1) s += __shfl_down(s, off, 32);
        float mu = __shfl(s, 0, 32) * (1.0f / DD);
        float var = 0.f;
        for (int d = lane; d < DD; d += 32) { float t = (float)row[d] - mu; var += t * t; }
#pragma unroll
        for (int off = 16; off > 0; off >>= 1) var += __shfl_down(var, off, 32);
        float rs = rsqrtf(__shfl(var, 0, 32) * (1.0f / DD) + 1e-5f);
        int j = jt * 16 + m;
        _Float16* dst = k16 + ((size_t)b * NKPAD + j) * DD;
        for (int d = lane; d < DD; d += 32) {
            float v = (j < NK) ? (((float)row[d] - mu) * rs * lnw[d] + lnb[d]) : 0.0f;
            dst[d] = (_Float16)v;
        }
    }
}

// ---------- kernel 4 (fused): unfold -> q_lin -> LN -> energy -> rowmax -> scores ----------
__global__ void __launch_bounds__(256) scores_kernel(
        const float* __restrict__ x, const _Float16* __restrict__ WqT,
        const float* __restrict__ bq, const float* __restrict__ lnw,
        const float* __restrict__ lnb, const _Float16* __restrict__ k16,
        float* __restrict__ scores) {
    __shared__ _Float16 sA[16 * DD];
    __shared__ _Float16 sC[16 * DD];
    __shared__ float smax[8][16];
    const int tid = threadIdx.x, wave = tid >> 5, lane = tid & 31;
    const int pt = blockIdx.x, b = blockIdx.y;

    // unfold gather: feature f -> (c = f/256, r = (f%256)/16, col = f%16)
    for (int m = 0; m < 16; ++m) {
        int p = pt * 16 + m;
        int pi = p / MS, pj = p % MS;
        for (int f = tid; f < DD; f += 256) {
            float v = 0.f;
            if (p < NPATCH) {
                int c = f >> 8, rem = f & 255, rr = rem >> 4, cc = rem & 15;
                v = x[(((size_t)b * 3 + c) * 224 + pi * 4 + rr) * 224 + (pj * 4 + cc)];
            }
            sA[m * DD + f] = (_Float16)v;
        }
    }
    __syncthreads();

    // GEMM1: q_lin tile (16x768) into sC
    for (int nt = wave; nt < NTILE; nt += 8) {
        v8f acc = {};
        const _Float16* bt = WqT + (size_t)nt * 16 * DD;
        for (int kc = 0; kc < DD; kc += 32)
            acc = wmma16(load_frag_a(sA + kc, DD, lane), load_frag_b(bt + kc, DD, lane), acc);
        const int half = lane >> 4, n = lane & 15;
        const float bias = bq[nt * 16 + n];
#pragma unroll
        for (int r = 0; r < 8; ++r)
            sC[(r + 8 * half) * DD + nt * 16 + n] = (_Float16)(acc[r] + bias);
    }
    __syncthreads();

    // LN(q_lin) -> sA (f16)
    for (int m = wave * 2; m < wave * 2 + 2; ++m) {
        const _Float16* row = sC + m * DD;
        float s = 0.f;
        for (int d = lane; d < DD; d += 32) s += (float)row[d];
#pragma unroll
        for (int off = 16; off > 0; off >>= 1) s += __shfl_down(s, off, 32);
        float mu = __shfl(s, 0, 32) * (1.0f / DD);
        float var = 0.f;
        for (int d = lane; d < DD; d += 32) { float t = (float)row[d] - mu; var += t * t; }
#pragma unroll
        for (int off = 16; off > 0; off >>= 1) var += __shfl_down(var, off, 32);
        float rs = rsqrtf(__shfl(var, 0, 32) * (1.0f / DD) + 1e-5f);
        for (int d = lane; d < DD; d += 32)
            sA[m * DD + d] = (_Float16)((((float)row[d]) - mu) * rs * lnw[d] + lnb[d]);
    }
    __syncthreads();

    // GEMM2: energy (16 x 196), fused row-max
    float rowmax[8];
#pragma unroll
    for (int r = 0; r < 8; ++r) rowmax[r] = -__builtin_inff();
    const _Float16* kb = k16 + (size_t)b * NKPAD * DD;
    for (int jt2 = wave; jt2 < 13; jt2 += 8) {
        v8f acc = {};
        const _Float16* bt = kb + (size_t)jt2 * 16 * DD;
        for (int kc = 0; kc < DD; kc += 32)
            acc = wmma16(load_frag_a(sA + kc, DD, lane), load_frag_b(bt + kc, DD, lane), acc);
        const int n = lane & 15;
        const bool valid = (jt2 * 16 + n) < NK;
#pragma unroll
        for (int r = 0; r < 8; ++r) {
            float v = valid ? acc[r] : -__builtin_inff();
#pragma unroll
            for (int off = 1; off < 16; off <<= 1) v = fmaxf(v, __shfl_xor(v, off, 32));
            rowmax[r] = fmaxf(rowmax[r], v);
        }
    }
    {
        const int half = lane >> 4, n = lane & 15;
        if (n == 0) {
#pragma unroll
            for (int r = 0; r < 8; ++r) smax[wave][r + 8 * half] = rowmax[r];
        }
    }
    __syncthreads();
    if (tid < 16) {
        float v = smax[0][tid];
        for (int w = 1; w < 8; ++w) v = fmaxf(v, smax[w][tid]);
        int p = pt * 16 + tid;
        if (p < NPATCH)
            scores[(size_t)b * NPATCH + p] = v * 0.03608439182435161f;  // 1/sqrt(768)
    }
}

// ---------- kernel 5: 3x3 NMS + iterative top-100 argmax per batch ----------
__global__ void __launch_bounds__(256) topk_kernel(
        const float* __restrict__ scores, int* __restrict__ topk) {
    __shared__ float ssc[NPATCH];
    __shared__ float sdiv[NPATCH];
    __shared__ float rv[256];
    __shared__ int   ri[256];
    const int tid = threadIdx.x, b = blockIdx.x;
    const float* sc = scores + (size_t)b * NPATCH;
    for (int i = tid; i < NPATCH; i += 256) ssc[i] = sc[i];
    __syncthreads();
    for (int idx = tid; idx < NPATCH; idx += 256) {
        int i = idx / MS, j = idx % MS;
        float s = ssc[idx], lm = -__builtin_inff();
        for (int di = -1; di <= 1; ++di) {
            int ii = i + di; if (ii < 0 || ii >= MS) continue;
            for (int dj = -1; dj <= 1; ++dj) {
                int jj = j + dj; if (jj < 0 || jj >= MS) continue;
                lm = fmaxf(lm, ssc[ii * MS + jj]);
            }
        }
        sdiv[idx] = (s == lm) ? s : 0.0f;
    }
    __syncthreads();
    for (int t = 0; t < NTOP; ++t) {
        float bv = -__builtin_inff(); int bi = 0x7fffffff;
        for (int idx = tid; idx < NPATCH; idx += 256) {
            float v = sdiv[idx];
            if (v > bv || (v == bv && idx < bi)) { bv = v; bi = idx; }
        }
        rv[tid] = bv; ri[tid] = bi;
        __syncthreads();
        for (int s = 128; s > 0; s >>= 1) {
            if (tid < s) {
                float ov = rv[tid + s]; int oi = ri[tid + s];
                if (ov > rv[tid] || (ov == rv[tid] && oi < ri[tid])) { rv[tid] = ov; ri[tid] = oi; }
            }
            __syncthreads();
        }
        if (tid == 0) { topk[b * NTOP + t] = ri[0]; sdiv[ri[0]] = -__builtin_inff(); }
        __syncthreads();
    }
}

// ---------- kernel 6: recompute q_lin for top-100, @ Wout + bout, scale, write out ----------
__global__ void __launch_bounds__(256) refined_kernel(
        const float* __restrict__ x, const _Float16* __restrict__ WqT,
        const float* __restrict__ bq, const _Float16* __restrict__ WoT,
        const float* __restrict__ bout, const float* __restrict__ gamma,
        const int* __restrict__ topk, float* __restrict__ out) {
    __shared__ _Float16 sA[16 * DD];
    __shared__ _Float16 sC[16 * DD];
    const int tid = threadIdx.x, wave = tid >> 5, lane = tid & 31;
    const int st = blockIdx.x, b = blockIdx.y;

    for (int m = 0; m < 16; ++m) {
        int t = st * 16 + m;
        int p = (t < NTOP) ? topk[b * NTOP + t] : 0;
        int pi = p / MS, pj = p % MS;
        for (int f = tid; f < DD; f += 256) {
            int c = f >> 8, rem = f & 255, rr = rem >> 4, cc = rem & 15;
            sA[m * DD + f] =
                (_Float16)x[(((size_t)b * 3 + c) * 224 + pi * 4 + rr) * 224 + (pj * 4 + cc)];
        }
    }
    __syncthreads();

    // GEMM1: q_lin of selected rows -> sC (f16, bias added)
    for (int nt = wave; nt < NTILE; nt += 8) {
        v8f acc = {};
        const _Float16* bt = WqT + (size_t)nt * 16 * DD;
        for (int kc = 0; kc < DD; kc += 32)
            acc = wmma16(load_frag_a(sA + kc, DD, lane), load_frag_b(bt + kc, DD, lane), acc);
        const int half = lane >> 4, n = lane & 15;
        const float bias = bq[nt * 16 + n];
#pragma unroll
        for (int r = 0; r < 8; ++r)
            sC[(r + 8 * half) * DD + nt * 16 + n] = (_Float16)(acc[r] + bias);
    }
    __syncthreads();

    // GEMM2: refined = q_lin_sel @ Wout + bout ; out row = gamma * refined
    const float g = gamma[0];
    for (int nt = wave; nt < NTILE; nt += 8) {
        v8f acc = {};
        const _Float16* bt = WoT + (size_t)nt * 16 * DD;
        for (int kc = 0; kc < DD; kc += 32)
            acc = wmma16(load_frag_a(sC + kc, DD, lane), load_frag_b(bt + kc, DD, lane), acc);
        const int half = lane >> 4, n = lane & 15;
        const float bias = bout[nt * 16 + n];
#pragma unroll
        for (int r = 0; r < 8; ++r) {
            int t = st * 16 + r + 8 * half;
            if (t < NTOP)
                out[((size_t)b * OUTROWS + 197 + t) * DD + nt * 16 + n] = g * (acc[r] + bias);
        }
    }
}

extern "C" void kernel_launch(void* const* d_in, const int* in_sizes, int n_in,
                              void* d_out, int out_size, void* d_ws, size_t ws_size,
                              hipStream_t stream) {
    const float* x    = (const float*)d_in[0];
    const float* vit  = (const float*)d_in[1];
    const float* Wq   = (const float*)d_in[2];
    const float* bq   = (const float*)d_in[3];
    const float* lnqw = (const float*)d_in[4];
    const float* lnqb = (const float*)d_in[5];
    const float* Wk   = (const float*)d_in[6];
    const float* bk   = (const float*)d_in[7];
    const float* lnkw = (const float*)d_in[8];
    const float* lnkb = (const float*)d_in[9];
    const float* Wo   = (const float*)d_in[10];
    const float* bout = (const float*)d_in[11];
    const float* gamma= (const float*)d_in[12];
    float* out = (float*)d_out;

    const int Bn = in_sizes[1] / (197 * DD);

    char* ws = (char*)d_ws;
    _Float16* WqT = (_Float16*)ws; ws += (size_t)DD * DD * sizeof(_Float16);
    _Float16* WkT = (_Float16*)ws; ws += (size_t)DD * DD * sizeof(_Float16);
    _Float16* WoT = (_Float16*)ws; ws += (size_t)DD * DD * sizeof(_Float16);
    _Float16* k16 = (_Float16*)ws; ws += (size_t)Bn * NKPAD * DD * sizeof(_Float16);
    float* scores = (float*)ws;    ws += (size_t)Bn * NPATCH * sizeof(float);
    int* topk     = (int*)ws;

    cvt_weights_kernel<<<dim3((DD * DD + 255) / 256, 3), 256, 0, stream>>>(
        Wq, Wk, Wo, WqT, WkT, WoT);

    copy_vit_kernel<<<(int)(((size_t)Bn * 197 * DD + 255) / 256), 256, 0, stream>>>(
        vit, out, Bn);

    kproj_kernel<<<dim3(13, Bn), 256, 0, stream>>>(vit, WkT, bk, lnkw, lnkb, k16);

    scores_kernel<<<dim3((NPATCH + 15) / 16, Bn), 256, 0, stream>>>(
        x, WqT, bq, lnqw, lnqb, k16, scores);

    topk_kernel<<<Bn, 256, 0, stream>>>(scores, topk);

    refined_kernel<<<dim3((NTOP + 15) / 16, Bn), 256, 0, stream>>>(
        x, WqT, bq, WoT, bout, gamma, topk, out);
}
